// GRU_teach_58162447122465
// MI455X (gfx1250) — matrix operand order repaired and checked by reading, not compiled, not run on hardware
//
#include <hip/hip_runtime.h>
#include <stdint.h>

// ---------------- CDNA5 WMMA types ----------------
typedef __attribute__((ext_vector_type(16))) __bf16 bf16x16;
typedef __attribute__((ext_vector_type(8)))  __bf16 bf16x8;
typedef __attribute__((ext_vector_type(8)))  float  f32x8;

// ---------------- problem constants ----------------
#define B_TOT 4096
#define XD    128      // X_DIM
#define YD    64       // Y_DIM
#define TT    200      // T
#define LATN  100      // LAT
#define LATP  112      // LAT padded to 7 tiles of 16
#define NG    336      // 3*LATP (r|z|n gate columns, tile-aligned)
#define KH    128      // padded K (h is 100 -> 128; x_il is exactly 128)
#define H1N   100
#define CHN   50

#define NWAVE   4
#define ROWS_WG (NWAVE*16)   // 64 batch rows per workgroup -> 64 workgroups

// ---------------- LDS layout (bytes) ----------------
#define OFF_WIH  0                         // [336][128] bf16 transposed  86016
#define OFF_WHH  86016                     // [336][128] bf16 transposed  86016
#define OFF_WL1  172032                    // [112][128] bf16 transposed  28672
#define OFF_WL2  200704                    // [ 64][128] bf16 transposed  16384
#define OFF_BIH  217088                    // [336] f32
#define OFF_BHH  218432                    // [336] f32
#define OFF_BL1  219776                    // [112] f32
#define OFF_BL2  220224                    // [ 64] f32
#define OFF_WAVE 220480
// per-wave scratch: sH [16][128] bf16 (4096) | sX [16][128] bf16 (4096)
//                   sYp [16][64] f32 (4096)  | sHf [16][104] f32 (6656)
#define WAVE_STRIDE 18944
#define SMEM_TOTAL  (OFF_WAVE + NWAVE*WAVE_STRIDE)   // 296256 B < 320 KB

// ---------------- WMMA helpers ----------------
__device__ __forceinline__ f32x8 wmma_bf16(bf16x16 a, bf16x16 b, f32x8 c) {
  // D = A(16x32 bf16) x B(32x16 bf16) + C(16x16 f32)
  return __builtin_amdgcn_wmma_f32_16x16x32_bf16(false, a, false, b,
                                                 (short)0, c, false, false);
}

// A fragment: buf is row-major [16 rows][KH] bf16.
// Lane L: row m=L&15, half=L>>4. elems 0..7 = A[m][kb+half*8+i],
// elems 8..15 = A[m][kb+16+half*8+i]  (ISA 16-bit A layout).
__device__ __forceinline__ bf16x16 a_frag(const __bf16* buf, int m, int half, int kb) {
  bf16x8 lo = *(const bf16x8*)(buf + m*KH + kb + half*8);
  bf16x8 hi = *(const bf16x8*)(buf + m*KH + kb + 16 + half*8);
  return __builtin_shufflevector(lo, hi, 0,1,2,3,4,5,6,7,8,9,10,11,12,13,14,15);
}

// B fragment: wt is transposed weight [n][KH] bf16 (wt[n][k] = W[k][n]).
// Lane L: col n, half=L>>4. elems 0..7 = B[kb+half*8+i][n], elems 8..15 = B[kb+16+half*8+i][n].
__device__ __forceinline__ bf16x16 b_frag(const __bf16* wt, int n, int half, int kb) {
  bf16x8 lo = *(const bf16x8*)(wt + n*KH + kb + half*8);
  bf16x8 hi = *(const bf16x8*)(wt + n*KH + kb + 16 + half*8);
  return __builtin_shufflevector(lo, hi, 0,1,2,3,4,5,6,7,8,9,10,11,12,13,14,15);
}

__device__ __forceinline__ float sigm(float v) { return 1.f/(1.f+__expf(-v)); }

// ---------------- kernel ----------------
extern "C" __global__ __launch_bounds__(128)
void gru_teach_wmma(const float* __restrict__ x,
                    const float* __restrict__ y,         // [B][YD][TT]
                    const uint8_t* __restrict__ ymask,   // [B][YD][TT] bool
                    const float* __restrict__ Wmu1, const float* __restrict__ bmu1,
                    const float* __restrict__ Wmu2, const float* __restrict__ bmu2,
                    const float* __restrict__ Wih,  const float* __restrict__ bih,
                    const float* __restrict__ Whh,  const float* __restrict__ bhh,
                    const float* __restrict__ Wl1,  const float* __restrict__ bl1,
                    const float* __restrict__ Wl2,  const float* __restrict__ bl2,
                    const float* __restrict__ Wc1,  const float* __restrict__ bc1,
                    const float* __restrict__ Wc2,  const float* __restrict__ bc2,
                    float* __restrict__ out,             // [B][YD][TT]
                    float* __restrict__ out_cls)         // [B]
{
  extern __shared__ char smem[];
  __bf16* sWih = (__bf16*)(smem + OFF_WIH);
  __bf16* sWhh = (__bf16*)(smem + OFF_WHH);
  __bf16* sWl1 = (__bf16*)(smem + OFF_WL1);
  __bf16* sWl2 = (__bf16*)(smem + OFF_WL2);
  float*  sBih = (float*)(smem + OFF_BIH);
  float*  sBhh = (float*)(smem + OFF_BHH);
  float*  sBl1 = (float*)(smem + OFF_BL1);
  float*  sBl2 = (float*)(smem + OFF_BL2);

  const int tid  = threadIdx.x;
  const int wave = tid >> 5;
  const int lane = tid & 31;
  const int m16  = lane & 15;   // B/C/D column within tile; A row within tile
  const int half = lane >> 4;

  char* wb = smem + OFF_WAVE + wave*WAVE_STRIDE;
  __bf16* sH  = (__bf16*)(wb);          // h staged for A-fragments
  __bf16* sX  = (__bf16*)(wb + 4096);   // x_il / o1 staged for A-fragments
  float*  sYp = (float*)(wb + 8192);    // previous prediction (teacher forcing)
  float*  sHf = (float*)(wb + 12288);   // f32 scratch [16][104]

  // ---- stage weights: f32 global -> bf16 LDS, transposed, zero-padded ----
  for (int idx = tid; idx < NG*KH; idx += 128) {
    int n = idx >> 7, k = idx & 127;
    int g = n / LATP, j = n - g*LATP;
    float vi = (j < LATN) ? Wih[k*300 + g*LATN + j] : 0.f;
    float vh = (j < LATN && k < LATN) ? Whh[k*300 + g*LATN + j] : 0.f;
    sWih[idx] = (__bf16)vi;
    sWhh[idx] = (__bf16)vh;
  }
  for (int idx = tid; idx < LATP*KH; idx += 128) {
    int n = idx >> 7, k = idx & 127;
    sWl1[idx] = (__bf16)((n < LATN && k < LATN) ? Wl1[k*LATN + n] : 0.f);
  }
  for (int idx = tid; idx < YD*KH; idx += 128) {
    int n = idx >> 7, k = idx & 127;
    sWl2[idx] = (__bf16)((k < LATN) ? Wl2[k*YD + n] : 0.f);
  }
  for (int n = tid; n < NG; n += 128) {
    int g = n / LATP, j = n - g*LATP;
    sBih[n] = (j < LATN) ? bih[g*LATN + j] : 0.f;
    sBhh[n] = (j < LATN) ? bhh[g*LATN + j] : 0.f;
  }
  for (int n = tid; n < LATP; n += 128) sBl1[n] = (n < LATN) ? bl1[n] : 0.f;
  for (int n = tid; n < YD;   n += 128) sBl2[n] = bl2[n];
  __syncthreads();

  const int r0 = blockIdx.x * ROWS_WG + wave*16;   // this wave's 16 batch rows

  // zero y_prev; zero the never-rewritten pad columns of sH (NaN safety)
  for (int i = lane; i < 16*YD; i += 32) sYp[i] = 0.f;
  for (int i = lane; i < 16*16; i += 32) sH[(i>>4)*KH + 112 + (i&15)] = (__bf16)0.f;

  // ---- h0 = tanh(x@Wmu1+bmu1)@Wmu2+bmu2 (one-time, scalar) ----
  for (int i = lane; i < 16*H1N; i += 32) {
    int m = i / H1N, j = i - m*H1N;
    const float* xr = x + (size_t)(r0+m)*XD;
    float s = bmu1[j];
    for (int k = 0; k < XD; ++k) s += xr[k]*Wmu1[k*H1N + j];
    sHf[m*104 + j] = tanhf(s);
  }
  // h held in registers in C/D fragment layout: h[nt][v] = h(row v+8*half, col nt*16+m16)
  f32x8 h[7];
#pragma unroll
  for (int nt = 0; nt < 7; ++nt) {
    int col = nt*16 + m16;
    f32x8 acc;
#pragma unroll
    for (int v = 0; v < 8; ++v) {
      float s = 0.f;
      if (col < LATN) {
        s = bmu2[col];
        const float* t1 = sHf + (v + 8*half)*104;
        for (int k = 0; k < H1N; ++k) s += t1[k]*Wmu2[k*LATN + col];
      }
      acc[v] = s;
    }
    h[nt] = acc;
  }
  // stage h0 as bf16 A-source
#pragma unroll
  for (int nt = 0; nt < 7; ++nt) {
    int col = nt*16 + m16;
#pragma unroll
    for (int v = 0; v < 8; ++v) sH[(v+8*half)*KH + col] = (__bf16)h[nt][v];
  }

  // ================= recurrent T loop (no cross-wave sync) =================
  for (int t = 0; t < TT; ++t) {
    // ---- build x_il = interleave(y_in, mask) in sX, bf16 [16][128] ----
#pragma unroll 4
    for (int m = 0; m < 16; ++m) {
      size_t base = ((size_t)(r0+m)*YD);
#pragma unroll
      for (int dd = 0; dd < 2; ++dd) {
        int d = lane + dd*32;
        size_t gi = (base + d)*TT + t;
        uint8_t mk = ymask[gi];
        float yv = y[gi];
        if (t + 1 < TT) { __builtin_prefetch(&y[gi+1], 0, 1); __builtin_prefetch(&ymask[gi+1], 0, 1); }
        float val = mk ? yv : sYp[m*YD + d];
        sX[m*KH + 2*d]     = (__bf16)val;
        sX[m*KH + 2*d + 1] = (__bf16)(mk ? 1.f : 0.f);
      }
    }

    // A fragments for this step (x_il and h), K = 128 -> 4 blocks of 32
    bf16x16 ax[4], ah[4];
#pragma unroll
    for (int kk = 0; kk < 4; ++kk) {
      ax[kk] = a_frag(sX, m16, half, kk*32);
      ah[kk] = a_frag(sH, m16, half, kk*32);
    }

    // ---- r,z gates: (gx+gh) tiles 0..13, fused accumulation ----
    f32x8 r[7], z[7];
#pragma unroll
    for (int nt = 0; nt < 14; ++nt) {
      int col = nt*16 + m16;
      float bsum = sBih[col] + sBhh[col];
      f32x8 acc;
#pragma unroll
      for (int v = 0; v < 8; ++v) acc[v] = bsum;
#pragma unroll
      for (int kk = 0; kk < 4; ++kk) {
        acc = wmma_bf16(ax[kk], b_frag(sWih, col, half, kk*32), acc);
        acc = wmma_bf16(ah[kk], b_frag(sWhh, col, half, kk*32), acc);
      }
      f32x8 sg;
#pragma unroll
      for (int v = 0; v < 8; ++v) sg[v] = sigm(acc[v]);
      if (nt < 7) r[nt] = sg; else z[nt-7] = sg;
    }

    // ---- n gate (gx_n, gh_n kept separate: n = tanh(gx_n + r*gh_n)) + h update ----
#pragma unroll
    for (int nt = 0; nt < 7; ++nt) {
      int col = (14+nt)*16 + m16;
      f32x8 accx, acch;
      float bx = sBih[col], bh = sBhh[col];
#pragma unroll
      for (int v = 0; v < 8; ++v) { accx[v] = bx; acch[v] = bh; }
#pragma unroll
      for (int kk = 0; kk < 4; ++kk) {
        accx = wmma_bf16(ax[kk], b_frag(sWih, col, half, kk*32), accx);
        acch = wmma_bf16(ah[kk], b_frag(sWhh, col, half, kk*32), acch);
      }
      f32x8 hv = h[nt], rv = r[nt], zv = z[nt], o;
#pragma unroll
      for (int v = 0; v < 8; ++v) {
        float nn = tanhf(accx[v] + rv[v]*acch[v]);
        o[v] = (1.f - zv[v])*nn + zv[v]*hv[v];
      }
      h[nt] = o;
    }

    // ---- stage new h (bf16) for next matmuls ----
#pragma unroll
    for (int nt = 0; nt < 7; ++nt) {
      int col = nt*16 + m16;
#pragma unroll
      for (int v = 0; v < 8; ++v) sH[(v+8*half)*KH + col] = (__bf16)h[nt][v];
    }

    // ---- o1 = tanh(h@Wl1 + bl1) -> reuse sX as A-source ----
    bf16x16 ah2[4];
#pragma unroll
    for (int kk = 0; kk < 4; ++kk) ah2[kk] = a_frag(sH, m16, half, kk*32);
#pragma unroll
    for (int nt = 0; nt < 7; ++nt) {
      int col = nt*16 + m16;
      f32x8 acc; float bb = sBl1[col];
#pragma unroll
      for (int v = 0; v < 8; ++v) acc[v] = bb;
#pragma unroll
      for (int kk = 0; kk < 4; ++kk)
        acc = wmma_bf16(ah2[kk], b_frag(sWl1, col, half, kk*32), acc);
#pragma unroll
      for (int v = 0; v < 8; ++v) sX[(v+8*half)*KH + col] = (__bf16)tanhf(acc[v]);
    }

    // ---- out_t = o1@Wl2 + bl2; write global + keep as next y_prev ----
    bf16x16 ao[4];
#pragma unroll
    for (int kk = 0; kk < 4; ++kk) ao[kk] = a_frag(sX, m16, half, kk*32);
#pragma unroll
    for (int nt = 0; nt < 4; ++nt) {
      int d = nt*16 + m16;
      f32x8 acc; float bb = sBl2[d];
#pragma unroll
      for (int v = 0; v < 8; ++v) acc[v] = bb;
#pragma unroll
      for (int kk = 0; kk < 4; ++kk)
        acc = wmma_bf16(ao[kk], b_frag(sWl2, d, half, kk*32), acc);
#pragma unroll
      for (int v = 0; v < 8; ++v) {
        int m = v + 8*half;
        float val = acc[v];
        sYp[m*YD + d] = val;
        __builtin_nontemporal_store(val, &out[((size_t)(r0+m)*YD + d)*TT + t]);
      }
    }
  } // ---- end T loop ----

  // ---- classifier head: sigmoid(relu(h_T@Wc1+bc1)@Wc2+bc2) ----
#pragma unroll
  for (int nt = 0; nt < 7; ++nt) {
    int col = nt*16 + m16;
    if (col < LATN) {
#pragma unroll
      for (int v = 0; v < 8; ++v) sHf[(v+8*half)*104 + col] = h[nt][v];
    }
  }
  if (lane < 16) {
    const float* hr = sHf + lane*104;
    float s2 = bc2[0];
    for (int ch = 0; ch < CHN; ++ch) {
      float s1 = bc1[ch];
      for (int k = 0; k < LATN; ++k) s1 += hr[k]*Wc1[k*CHN + ch];
      s1 = s1 > 0.f ? s1 : 0.f;
      s2 += s1*Wc2[ch];
    }
    out_cls[r0 + lane] = sigm(s2);
  }
}

extern "C" void kernel_launch(void* const* d_in, const int* in_sizes, int n_in,
                              void* d_out, int out_size, void* d_ws, size_t ws_size,
                              hipStream_t stream) {
  (void)in_sizes; (void)n_in; (void)d_ws; (void)ws_size; (void)out_size;
  const float*   x    = (const float*)d_in[0];
  const float*   y    = (const float*)d_in[1];
  const uint8_t* ym   = (const uint8_t*)d_in[2];
  const float *Wmu1 = (const float*)d_in[3],  *bmu1 = (const float*)d_in[4];
  const float *Wmu2 = (const float*)d_in[5],  *bmu2 = (const float*)d_in[6];
  // d_in[7..10]: Wsig1/bsig1/Wsig2/bsig2 unused in eval-mode reference
  const float *Wih  = (const float*)d_in[11], *bih  = (const float*)d_in[12];
  const float *Whh  = (const float*)d_in[13], *bhh  = (const float*)d_in[14];
  const float *Wl1  = (const float*)d_in[15], *bl1  = (const float*)d_in[16];
  const float *Wl2  = (const float*)d_in[17], *bl2  = (const float*)d_in[18];
  const float *Wc1  = (const float*)d_in[19], *bc1  = (const float*)d_in[20];
  const float *Wc2  = (const float*)d_in[21], *bc2  = (const float*)d_in[22];

  float* out  = (float*)d_out;                       // [4096][64][200]
  float* ocls = out + (size_t)B_TOT*YD*TT;           // [4096]

  static_assert(SMEM_TOTAL <= 320*1024, "LDS budget");
  (void)hipFuncSetAttribute((const void*)gru_teach_wmma,
                            hipFuncAttributeMaxDynamicSharedMemorySize, SMEM_TOTAL);

  dim3 grid(B_TOT / ROWS_WG);   // 64 workgroups
  dim3 block(128);              // 4 wave32s, each owns 16 batch rows
  gru_teach_wmma<<<grid, block, SMEM_TOTAL, stream>>>(
      x, y, ym, Wmu1, bmu1, Wmu2, bmu2,
      Wih, bih, Whh, bhh, Wl1, bl1, Wl2, bl2,
      Wc1, bc1, Wc2, bc2, out, ocls);
}
